// GPT2Attention_17042430231396
// MI455X (gfx1250) — matrix operand order repaired and checked by reading, not compiled, not run on hardware
//
#include <hip/hip_runtime.h>
#include <hip/hip_bf16.h>

// ---------------------------------------------------------------------------
// Types for CDNA5 WMMA (wave32): v_wmma_f32_16x16x32_bf16
// ---------------------------------------------------------------------------
typedef __bf16 bf16;
typedef bf16  v16bf __attribute__((ext_vector_type(16)));
typedef bf16  v8bf  __attribute__((ext_vector_type(8)));
typedef float v8f   __attribute__((ext_vector_type(8)));

union BF16x16 {
    v16bf v;
    struct { v8bf lo, hi; } h;
};

// fp32 -> bf16 round-to-nearest-even
static __device__ __forceinline__ bf16 f2bf(float f) {
    unsigned u = __builtin_bit_cast(unsigned, f);
    unsigned r = (u + 0x7FFFu + ((u >> 16) & 1u)) >> 16;
    unsigned short s = (unsigned short)r;
    return __builtin_bit_cast(bf16, s);
}

static __device__ __forceinline__ unsigned pack2(bf16 a, bf16 b) {
    unsigned lo = __builtin_bit_cast(unsigned short, a);
    unsigned hi = __builtin_bit_cast(unsigned short, b);
    return lo | (hi << 16);
}

// Async global -> LDS copy (gfx1250). Tracked by ASYNCcnt.
static __device__ __forceinline__ void async_load_b128(unsigned lds_off,
                                                       const void* gptr) {
    unsigned long long ga = (unsigned long long)gptr;
    asm volatile("global_load_async_to_lds_b128 %0, %1, off"
                 :: "v"(lds_off), "v"(ga)
                 : "memory");
}
static __device__ __forceinline__ void wait_async0() {
    asm volatile("s_wait_asynccnt 0" ::: "memory");
}

// ---------------------------------------------------------------------------
// fp32 -> bf16 conversion kernel
// ---------------------------------------------------------------------------
__global__ __launch_bounds__(256) void cvt_f32_bf16(const float* __restrict__ in,
                                                    bf16* __restrict__ out,
                                                    int n) {
    int i = (blockIdx.x * 256 + threadIdx.x) * 4;
    if (i + 3 < n) {
        float4 v = *reinterpret_cast<const float4*>(in + i);
        out[i + 0] = f2bf(v.x);
        out[i + 1] = f2bf(v.y);
        out[i + 2] = f2bf(v.z);
        out[i + 3] = f2bf(v.w);
    }
}

// ---------------------------------------------------------------------------
// WMMA bf16 GEMM: C[M,N] = A[M,K] * B[K,N] + bias
// block = 256 threads = 8 waves; block tile 128x128; wave tile 32x64; BK=64.
// Double-buffered LDS: async-to-LDS prefetch of tile k+1 overlaps the 16
// WMMAs on tile k; s_wait_asynccnt + barrier close each stage.
// ---------------------------------------------------------------------------
template <bool OUT_BF16>
__global__ __launch_bounds__(256) void gemm_bf16_wmma(
    const bf16* __restrict__ A, const bf16* __restrict__ Bm,
    const float* __restrict__ bias, void* __restrict__ C,
    int M, int N, int K) {
    __shared__ bf16 As[2][128 * 64];   // 2 x 16 KB, row-major [m][k]
    __shared__ bf16 Bt[2][128 * 72];   // 2 x 18 KB, transposed [n][k], pad 72

    const int tid  = threadIdx.x;
    const int wave = tid >> 5;
    const int lane = tid & 31;
    const int l16  = lane & 15;
    const int lh   = lane >> 4;        // half-wave select
    const int wm   = wave >> 1;        // 0..3
    const int wn   = wave & 1;         // 0..1
    const int m0   = blockIdx.y * 128;
    const int n0   = blockIdx.x * 128;

    // ---- tile stage: async A copy + transposed (paired-b32) B stores ----
    auto stage = [&](int buf, int k0) {
        // A tile 128x64: 1024 b128 chunks, 4 async per thread
        #pragma unroll
        for (int cc = 0; cc < 4; ++cc) {
            int c    = tid + cc * 256;       // 0..1023
            int row  = c >> 3;
            int col8 = (c & 7) * 8;
            unsigned loff = (unsigned)(size_t)&As[buf][row * 64 + col8];
            async_load_b128(loff, A + (size_t)(m0 + row) * K + k0 + col8);
        }
        // B tile 64x128 -> Bt[n][k]; handle k-pairs, store packed b32
        #pragma unroll
        for (int p = 0; p < 2; ++p) {
            int idx = tid + p * 256;         // 0..511
            int k   = (idx >> 4) * 2;        // even k
            int nc  = (idx & 15) * 8;
            const bf16* g0 = Bm + (size_t)(k0 + k) * N + n0 + nc;
            v8bf d0 = *reinterpret_cast<const v8bf*>(g0);
            v8bf d1 = *reinterpret_cast<const v8bf*>(g0 + N);
            #pragma unroll
            for (int i = 0; i < 8; ++i)
                *reinterpret_cast<unsigned*>(&Bt[buf][(nc + i) * 72 + k]) =
                    pack2(d0[i], d1[i]);
        }
    };

    v8f acc[2][4] = {};

    stage(0, 0);
    wait_async0();
    __syncthreads();

    const int nsteps = K >> 6;            // K / 64
    for (int s = 0; s < nsteps; ++s) {
        const int cur = s & 1, nxt = cur ^ 1;
        if (s + 1 < nsteps) stage(nxt, (s + 1) << 6);   // async prefetch

        // ---- compute on current tile: 2 K-chunks of 32, 8 WMMAs each ----
        const int hk = 8 * lh;
        #pragma unroll
        for (int c = 0; c < 2; ++c) {
            BF16x16 afr[2], bfr[4];
            #pragma unroll
            for (int tm = 0; tm < 2; ++tm) {
                int r = wm * 32 + tm * 16 + l16;
                afr[tm].h.lo = *reinterpret_cast<const v8bf*>(
                    &As[cur][r * 64 + c * 32 + hk]);
                afr[tm].h.hi = *reinterpret_cast<const v8bf*>(
                    &As[cur][r * 64 + c * 32 + 16 + hk]);
            }
            #pragma unroll
            for (int tn = 0; tn < 4; ++tn) {
                int n = wn * 64 + tn * 16 + l16;
                bfr[tn].h.lo = *reinterpret_cast<const v8bf*>(
                    &Bt[cur][n * 72 + c * 32 + hk]);
                bfr[tn].h.hi = *reinterpret_cast<const v8bf*>(
                    &Bt[cur][n * 72 + c * 32 + 16 + hk]);
            }
            #pragma unroll
            for (int tm = 0; tm < 2; ++tm)
                #pragma unroll
                for (int tn = 0; tn < 4; ++tn)
                    acc[tm][tn] = __builtin_amdgcn_wmma_f32_16x16x32_bf16(
                        false, afr[tm].v, false, bfr[tn].v,
                        (short)0, acc[tm][tn], false, false);
        }

        wait_async0();        // next tile's async A data landed
        __syncthreads();      // all waves done reading cur + B stores visible
    }

    // --- epilogue: bias + store ---
    #pragma unroll
    for (int tm = 0; tm < 2; ++tm) {
        #pragma unroll
        for (int tn = 0; tn < 4; ++tn) {
            int n = n0 + wn * 64 + tn * 16 + l16;
            float bv = bias[n];
            #pragma unroll
            for (int i = 0; i < 8; ++i) {
                int m = m0 + wm * 32 + tm * 16 + i + 8 * lh;
                float v = acc[tm][tn][i] + bv;
                if (OUT_BF16)
                    ((bf16*)C)[(size_t)m * N + n] = f2bf(v);
                else
                    ((float*)C)[(size_t)m * N + n] = v;
            }
        }
    }
}

// ---------------------------------------------------------------------------
// Flash-style causal attention, bf16 WMMA, f32 online softmax.
// qkv: [B, S, 3*D] bf16 (Q at col 0, K at col D, V at col 2D; head h uses
// cols h*64..h*64+63 of each). attn: [B, S, D] bf16.
// grid = (S/128, B*H), block = 256 (8 waves, 16 q-rows per wave).
// ---------------------------------------------------------------------------
__global__ __launch_bounds__(256) void attn_wmma(const bf16* __restrict__ qkv,
                                                 bf16* __restrict__ attn) {
    constexpr int S = 2048, D = 1024, D3 = 3072, HD = 64, H = 16;
    __shared__ bf16 VT[64 * 72];        // V^T tile: [d][kv], padded stride 72
    __shared__ bf16 PL[8][16 * 64];     // per-wave P tile (C-layout dump)

    const int tid  = threadIdx.x;
    const int wave = tid >> 5;
    const int lane = tid & 31;
    const int l16  = lane & 15;
    const int lh   = lane >> 4;
    const int b    = blockIdx.y >> 4;
    const int h    = blockIdx.y & (H - 1);
    const bf16* qb = qkv + (size_t)b * S * D3;
    const int q0   = blockIdx.x * 128 + wave * 16;
    const int hk   = 8 * lh;

    // Q fragments for this wave's 16 rows (K-dim = HD = 64 -> 2 chunks of 32)
    BF16x16 qa[2];
    {
        const bf16* qp = qb + (size_t)(q0 + l16) * D3 + h * HD;
        #pragma unroll
        for (int c = 0; c < 2; ++c) {
            qa[c].h.lo = *reinterpret_cast<const v8bf*>(qp + c * 32 + hk);
            qa[c].h.hi = *reinterpret_cast<const v8bf*>(qp + c * 32 + 16 + hk);
        }
    }

    v8f o[4] = {};
    float mrow[8], lrow[8];
    #pragma unroll
    for (int i = 0; i < 8; ++i) { mrow[i] = -3.0e38f; lrow[i] = 0.0f; }

    const int jtmax = blockIdx.x * 2 + 1;
    for (int jt = 0; jt <= jtmax; ++jt) {
        const int kv0 = jt * 64;
        __syncthreads();
        // --- block-cooperative V^T staging: 64 kv-rows x 64 d ---
        {
            int kvl = tid >> 2;                 // 0..63
            int d0  = (tid & 3) * 16;
            const bf16* vp =
                qb + (size_t)(kv0 + kvl) * D3 + 2 * D + h * HD + d0;
            v8bf v0 = *reinterpret_cast<const v8bf*>(vp);
            v8bf v1 = *reinterpret_cast<const v8bf*>(vp + 8);
            #pragma unroll
            for (int i = 0; i < 8; ++i) {
                VT[(d0 + i)     * 72 + kvl] = v0[i];
                VT[(d0 + 8 + i) * 72 + kvl] = v1[i];
            }
        }
        __syncthreads();
        if (kv0 > q0 + 15) continue;   // fully masked for this wave

        // --- S = Q K^T (16 x 64), 4 col-tiles x 2 K-chunks ---
        v8f sacc[4] = {};
        #pragma unroll
        for (int tn = 0; tn < 4; ++tn) {
            #pragma unroll
            for (int c = 0; c < 2; ++c) {
                BF16x16 kb;
                const bf16* kp = qb + (size_t)(kv0 + tn * 16 + l16) * D3 + D +
                                 h * HD + c * 32 + hk;
                kb.h.lo = *reinterpret_cast<const v8bf*>(kp);
                kb.h.hi = *reinterpret_cast<const v8bf*>(kp + 16);
                sacc[tn] = __builtin_amdgcn_wmma_f32_16x16x32_bf16(
                    false, qa[c].v, false, kb.v, (short)0, sacc[tn],
                    false, false);
            }
        }

        // --- scale + causal mask + row max ---
        float newm[8];
        #pragma unroll
        for (int i = 0; i < 8; ++i) newm[i] = mrow[i];
        #pragma unroll
        for (int tn = 0; tn < 4; ++tn) {
            int col = kv0 + tn * 16 + l16;
            #pragma unroll
            for (int i = 0; i < 8; ++i) {
                int row = q0 + i + 8 * lh;
                float v = sacc[tn][i] * 0.125f;        // 1/sqrt(64)
                v = (col <= row) ? v : -3.0e38f;
                sacc[tn][i] = v;
                newm[i] = fmaxf(newm[i], v);
            }
        }
        #pragma unroll
        for (int i = 0; i < 8; ++i) {                  // reduce over 16-lane half
            float v = newm[i];
            v = fmaxf(v, __shfl_xor(v, 1, 32));
            v = fmaxf(v, __shfl_xor(v, 2, 32));
            v = fmaxf(v, __shfl_xor(v, 4, 32));
            v = fmaxf(v, __shfl_xor(v, 8, 32));
            newm[i] = v;
        }
        float alpha[8];
        #pragma unroll
        for (int i = 0; i < 8; ++i) {
            alpha[i] = __expf(mrow[i] - newm[i]);
            mrow[i]  = newm[i];
        }
        float rs[8] = {};
        #pragma unroll
        for (int tn = 0; tn < 4; ++tn)
            #pragma unroll
            for (int i = 0; i < 8; ++i) {
                float p = __expf(sacc[tn][i] - newm[i]);
                sacc[tn][i] = p;
                rs[i] += p;
            }
        #pragma unroll
        for (int i = 0; i < 8; ++i) {
            float v = rs[i];
            v += __shfl_xor(v, 1, 32);
            v += __shfl_xor(v, 2, 32);
            v += __shfl_xor(v, 4, 32);
            v += __shfl_xor(v, 8, 32);
            lrow[i] = lrow[i] * alpha[i] + v;
        }
        #pragma unroll
        for (int t = 0; t < 4; ++t)
            #pragma unroll
            for (int i = 0; i < 8; ++i) o[t][i] *= alpha[i];

        // --- P (C-layout) -> wave LDS -> A-layout fragments ---
        bf16* pl = &PL[wave][0];
        #pragma unroll
        for (int tn = 0; tn < 4; ++tn)
            #pragma unroll
            for (int i = 0; i < 8; ++i)
                pl[(i + 8 * lh) * 64 + tn * 16 + l16] = f2bf(sacc[tn][i]);
        BF16x16 pa[2];
        #pragma unroll
        for (int c = 0; c < 2; ++c) {
            pa[c].h.lo =
                *reinterpret_cast<const v8bf*>(&pl[l16 * 64 + c * 32 + hk]);
            pa[c].h.hi =
                *reinterpret_cast<const v8bf*>(&pl[l16 * 64 + c * 32 + 16 + hk]);
        }

        // --- O += P * V  (B fragments from V^T tile in LDS) ---
        #pragma unroll
        for (int t = 0; t < 4; ++t) {
            #pragma unroll
            for (int c = 0; c < 2; ++c) {
                BF16x16 vb;
                const bf16* vt = &VT[(t * 16 + l16) * 72 + c * 32 + hk];
                vb.h.lo = *reinterpret_cast<const v8bf*>(vt);
                vb.h.hi = *reinterpret_cast<const v8bf*>(vt + 16);
                o[t] = __builtin_amdgcn_wmma_f32_16x16x32_bf16(
                    false, pa[c].v, false, vb.v, (short)0, o[t], false, false);
            }
        }
    }

    // --- epilogue: O /= l, store to attn[b][s][h*64 + d] ---
    #pragma unroll
    for (int i = 0; i < 8; ++i) lrow[i] = 1.0f / lrow[i];
    bf16* ap = attn + (size_t)b * S * D;
    #pragma unroll
    for (int t = 0; t < 4; ++t)
        #pragma unroll
        for (int i = 0; i < 8; ++i) {
            int s = q0 + i + 8 * lh;
            ap[(size_t)s * D + h * HD + t * 16 + l16] = f2bf(o[t][i] * lrow[i]);
        }
}

// ---------------------------------------------------------------------------
// Host-side launch
// ---------------------------------------------------------------------------
extern "C" void kernel_launch(void* const* d_in, const int* in_sizes, int n_in,
                              void* d_out, int out_size, void* d_ws,
                              size_t ws_size, hipStream_t stream) {
    constexpr int B = 2, S = 2048, D = 1024, H = 16;
    constexpr int M = B * S;            // 4096
    constexpr int N3 = 3 * D;           // 3072

    const float* x        = (const float*)d_in[0];
    const float* c_attn_w = (const float*)d_in[1];
    const float* c_attn_b = (const float*)d_in[2];
    const float* c_proj_w = (const float*)d_in[3];
    const float* c_proj_b = (const float*)d_in[4];
    float* out = (float*)d_out;

    char* ws = (char*)d_ws;
    size_t off = 0;
    bf16* xb    = (bf16*)(ws + off); off += (size_t)M * D * 2;     // 8 MB
    bf16* wqkv  = (bf16*)(ws + off); off += (size_t)D * N3 * 2;    // 6 MB
    bf16* wproj = (bf16*)(ws + off); off += (size_t)D * D * 2;     // 2 MB
    bf16* qkvb  = (bf16*)(ws + off); off += (size_t)M * N3 * 2;    // 24 MB
    bf16* attnb = (bf16*)(ws + off); off += (size_t)M * D * 2;     // 8 MB
    (void)ws_size; (void)n_in; (void)in_sizes; (void)out_size;

    // fp32 -> bf16 conversions
    cvt_f32_bf16<<<(M * D / 4 + 255) / 256, 256, 0, stream>>>(x, xb, M * D);
    cvt_f32_bf16<<<(D * N3 / 4 + 255) / 256, 256, 0, stream>>>(c_attn_w, wqkv,
                                                               D * N3);
    cvt_f32_bf16<<<(D * D / 4 + 255) / 256, 256, 0, stream>>>(c_proj_w, wproj,
                                                              D * D);

    // QKV projection: [4096,1024] x [1024,3072] + bias -> bf16
    dim3 g1(N3 / 128, M / 128);
    gemm_bf16_wmma<true><<<g1, 256, 0, stream>>>(xb, wqkv, c_attn_b, qkvb,
                                                 M, N3, D);

    // causal flash attention
    dim3 g2(S / 128, B * H);
    attn_wmma<<<g2, 256, 0, stream>>>(qkvb, attnb);

    // output projection: [4096,1024] x [1024,1024] + bias -> fp32 out
    dim3 g3(D / 128, M / 128);
    gemm_bf16_wmma<false><<<g3, 256, 0, stream>>>(attnb, wproj, c_proj_b, out,
                                                  M, D, D);
}
